// BahdanauAttention_38895223832780
// MI455X (gfx1250) — matrix operand order repaired and checked by reading, not compiled, not run on hardware
//
#include <hip/hip_runtime.h>
#include <hip/hip_bf16.h>

#define B_DIM   32
#define S_DIM   4096
#define ENC_DIM 512
#define DEC_DIM 512

#define ROWS    64              // s-rows per block (4 M-tiles of 16)
#define MT      4               // M-tiles per block
#define LDS_STRIDE 520          // padded bf16 row stride (breaks bank conflicts)

typedef __attribute__((ext_vector_type(16))) __bf16 v16bf;
typedef __attribute__((ext_vector_type(8)))  float  v8f;

union FragBF {
  v16bf v;
  unsigned int u[8];
};

__device__ __forceinline__ unsigned short f32_to_bf16_rne(float f) {
  unsigned int u = __float_as_uint(f);
  u += 0x7fffu + ((u >> 16) & 1u);   // round-to-nearest-even
  return (unsigned short)(u >> 16);
}

// ---------------------------------------------------------------------------
// Kernel 1: convert W_e = W_attn[:, 512:] (f32, row stride 1024) -> bf16 Wt[d][e]
// ---------------------------------------------------------------------------
__global__ __launch_bounds__(256) void cvt_we_kernel(const float* __restrict__ W,
                                                     unsigned short* __restrict__ Wt) {
  int idx = blockIdx.x * 256 + threadIdx.x;          // 0 .. 512*512-1
  int d = idx >> 9;
  int e = idx & 511;
  Wt[idx] = f32_to_bf16_rne(W[(size_t)d * (ENC_DIM + DEC_DIM) + DEC_DIM + e]);
}

// ---------------------------------------------------------------------------
// Kernel 2: hb[b,d] = b_attn[d] + sum_k dh[b,k] * W_attn[d,k]
// ---------------------------------------------------------------------------
__global__ __launch_bounds__(256) void hb_kernel(const float* __restrict__ dh,
                                                 const float* __restrict__ W,
                                                 const float* __restrict__ bias,
                                                 float* __restrict__ hb) {
  int idx = blockIdx.x * 256 + threadIdx.x;          // 0 .. 32*512-1
  int b = idx >> 9;
  int d = idx & 511;
  const float* w = W + (size_t)d * (ENC_DIM + DEC_DIM);
  const float* h = dh + (size_t)b * DEC_DIM;
  float acc = bias[d];
  #pragma unroll 8
  for (int e = 0; e < DEC_DIM; ++e) acc = fmaf(h[e], w[e], acc);
  hb[idx] = acc;
}

// ---------------------------------------------------------------------------
// Kernel 3: scores[b, s0:s0+64] = v . tanh(hb[b,:] + enc_tile @ We^T)
// Block: 256 threads (8 waves), 64 s-rows (4 M-tiles), all 512 d.
// Wave w handles d-tile pairs {w, w+8} then {w+16, w+24}. Inner k-step:
// 4 A-frags (LDS, reused across 2 d-tiles) x 2 B-frags -> 8 WMMAs.
// Dynamic LDS: A tile (64 x 520 bf16) + 8x64 f32 reduction buffer.
// ---------------------------------------------------------------------------
__global__ __launch_bounds__(256) void attn_scores_kernel(
    const float* __restrict__ enc,            // (B, S, ENC) f32
    const unsigned short* __restrict__ Wt,    // (DEC, ENC) bf16
    const float* __restrict__ hb,             // (B, DEC)
    const float* __restrict__ vvec,           // (DEC)
    float* __restrict__ scores) {             // (B, S)
  extern __shared__ char dynsm[];
  unsigned short* As = (unsigned short*)dynsm;                       // 64*520 bf16
  float* red = (float*)(dynsm + ROWS * LDS_STRIDE * sizeof(unsigned short)); // [8][64]

  const int tid = threadIdx.x;
  const int b   = blockIdx.y;
  const int s0  = blockIdx.x * ROWS;

  // ---- stage + convert the 64x512 f32 A tile into LDS as bf16 ----
  const float* encb = enc + ((size_t)b * S_DIM + s0) * ENC_DIM;
  for (int idx = tid; idx < ROWS * 128; idx += 256) {   // float4 granularity
    int r = idx >> 7;
    int e = (idx & 127) << 2;
    float4 f = *reinterpret_cast<const float4*>(encb + r * ENC_DIM + e);
    unsigned int lo = (unsigned int)f32_to_bf16_rne(f.x) |
                      ((unsigned int)f32_to_bf16_rne(f.y) << 16);
    unsigned int hi = (unsigned int)f32_to_bf16_rne(f.z) |
                      ((unsigned int)f32_to_bf16_rne(f.w) << 16);
    unsigned long long p = (unsigned long long)lo | ((unsigned long long)hi << 32);
    *reinterpret_cast<unsigned long long*>(&As[r * LDS_STRIDE + e]) = p;
  }
  __syncthreads();

  const int wave = tid >> 5;
  const int lane = tid & 31;
  const int half = lane >> 4;   // 16-lane half
  const int lrow = lane & 15;   // M row (A) / N column (B, C/D)

  float part[MT][8];
  #pragma unroll
  for (int m = 0; m < MT; ++m)
    #pragma unroll
    for (int i = 0; i < 8; ++i) part[m][i] = 0.0f;

  #pragma unroll
  for (int p = 0; p < 2; ++p) {
    const int dt0 = wave + p * 16;       // d-tile pair: dt0 and dt0+8
    const int dc0 = dt0 * 16 + lrow;
    const int dc1 = dc0 + 128;           // (dt0+8)*16 + lrow
    const unsigned short* wrow0 = Wt + (size_t)dc0 * ENC_DIM;
    const unsigned short* wrow1 = Wt + (size_t)dc1 * ENC_DIM;

    v8f c[2][MT];
    #pragma unroll
    for (int j = 0; j < 2; ++j)
      #pragma unroll
      for (int m = 0; m < MT; ++m) c[j][m] = v8f{0.f,0.f,0.f,0.f,0.f,0.f,0.f,0.f};

    #pragma unroll 2
    for (int k0 = 0; k0 < ENC_DIM; k0 += 32) {
      // B fragments (32x16 bf16): half 0 -> K {k0..k0+15}, half 1 -> {k0+16..31}
      FragBF bm0, bm1;
      {
        const unsigned short* bp = wrow0 + k0 + half * 16;
        uint4 w0 = *reinterpret_cast<const uint4*>(bp);
        uint4 w1 = *reinterpret_cast<const uint4*>(bp + 8);
        bm0.u[0] = w0.x; bm0.u[1] = w0.y; bm0.u[2] = w0.z; bm0.u[3] = w0.w;
        bm0.u[4] = w1.x; bm0.u[5] = w1.y; bm0.u[6] = w1.z; bm0.u[7] = w1.w;
      }
      {
        const unsigned short* bp = wrow1 + k0 + half * 16;
        uint4 w0 = *reinterpret_cast<const uint4*>(bp);
        uint4 w1 = *reinterpret_cast<const uint4*>(bp + 8);
        bm1.u[0] = w0.x; bm1.u[1] = w0.y; bm1.u[2] = w0.z; bm1.u[3] = w0.w;
        bm1.u[4] = w1.x; bm1.u[5] = w1.y; bm1.u[6] = w1.z; bm1.u[7] = w1.w;
      }
      __builtin_prefetch(wrow0 + k0 + 64, 0, 1);
      __builtin_prefetch(wrow1 + k0 + 64, 0, 1);

      // A fragments (16x32 bf16): half 0 -> K {k0+0..15},
      //                           half 1 -> K {k0+16..31}
      #pragma unroll
      for (int m = 0; m < MT; ++m) {
        FragBF a;
        const unsigned short* ap = &As[(m * 16 + lrow) * LDS_STRIDE + k0 + half * 16];
        uint4 t0 = *reinterpret_cast<const uint4*>(ap);
        uint4 t1 = *reinterpret_cast<const uint4*>(ap + 8);
        a.u[0] = t0.x; a.u[1] = t0.y; a.u[2] = t0.z; a.u[3] = t0.w;
        a.u[4] = t1.x; a.u[5] = t1.y; a.u[6] = t1.z; a.u[7] = t1.w;

        c[0][m] = __builtin_amdgcn_wmma_f32_16x16x32_bf16(
                      false, a.v, false, bm0.v, (short)0, c[0][m], false, false);
        c[1][m] = __builtin_amdgcn_wmma_f32_16x16x32_bf16(
                      false, a.v, false, bm1.v, (short)0, c[1][m], false, false);
      }
    }

    // epilogue for this d-tile pair: tanh(c + hb) * v into per-row partials
    #pragma unroll
    for (int j = 0; j < 2; ++j) {
      const int dcol = (j == 0) ? dc0 : dc1;
      const float hbv = hb[(size_t)b * DEC_DIM + dcol];
      const float vv  = vvec[dcol];
      #pragma unroll
      for (int m = 0; m < MT; ++m)
        #pragma unroll
        for (int i = 0; i < 8; ++i)
          part[m][i] += tanhf(c[j][m][i] + hbv) * vv;
    }
  }

  // reduce across the 16 lanes of each half (C/D layout: VGPR i holds row i for
  // lanes 0-15 and row i+8 for lanes 16-31; lane = d column)
  #pragma unroll
  for (int msk = 1; msk < 16; msk <<= 1) {
    #pragma unroll
    for (int m = 0; m < MT; ++m)
      #pragma unroll
      for (int i = 0; i < 8; ++i)
        part[m][i] += __shfl_xor(part[m][i], msk, 32);
  }

  if (lrow == 0) {
    #pragma unroll
    for (int m = 0; m < MT; ++m)
      #pragma unroll
      for (int i = 0; i < 8; ++i)
        red[wave * ROWS + m * 16 + half * 8 + i] = part[m][i];
  }
  __syncthreads();

  if (tid < ROWS) {
    float s = 0.0f;
    #pragma unroll
    for (int w = 0; w < 8; ++w) s += red[w * ROWS + tid];
    scores[(size_t)b * S_DIM + s0 + tid] = s;
  }
}

// ---------------------------------------------------------------------------
// Kernel 4: in-place softmax over S per batch row
// ---------------------------------------------------------------------------
__global__ __launch_bounds__(256) void softmax_kernel(float* __restrict__ out) {
  const int b = blockIdx.x;
  float* row = out + (size_t)b * S_DIM;
  const int tid = threadIdx.x;
  __shared__ float sm[256];

  float x[16];
  float lmax = -3.402823466e38f;
  #pragma unroll
  for (int i = 0; i < 16; ++i) {
    x[i] = row[tid + i * 256];
    lmax = fmaxf(lmax, x[i]);
  }
  sm[tid] = lmax;
  __syncthreads();
  for (int s = 128; s > 0; s >>= 1) {
    if (tid < s) sm[tid] = fmaxf(sm[tid], sm[tid + s]);
    __syncthreads();
  }
  const float mx = sm[0];
  __syncthreads();

  float lsum = 0.0f;
  #pragma unroll
  for (int i = 0; i < 16; ++i) {
    x[i] = __expf(x[i] - mx);
    lsum += x[i];
  }
  sm[tid] = lsum;
  __syncthreads();
  for (int s = 128; s > 0; s >>= 1) {
    if (tid < s) sm[tid] += sm[tid + s];
    __syncthreads();
  }
  const float inv = 1.0f / sm[0];
  #pragma unroll
  for (int i = 0; i < 16; ++i) row[tid + i * 256] = x[i] * inv;
}

// ---------------------------------------------------------------------------
extern "C" void kernel_launch(void* const* d_in, const int* in_sizes, int n_in,
                              void* d_out, int out_size, void* d_ws, size_t ws_size,
                              hipStream_t stream) {
  (void)in_sizes; (void)n_in; (void)out_size; (void)ws_size;

  const float* dh   = (const float*)d_in[0];   // (B, DEC)
  const float* enc  = (const float*)d_in[1];   // (B, S, ENC)
  const float* W    = (const float*)d_in[2];   // (DEC, ENC+DEC)
  const float* bias = (const float*)d_in[3];   // (DEC,)
  const float* v    = (const float*)d_in[4];   // (DEC,)
  float* out = (float*)d_out;                  // (B, S)

  unsigned short* Wt = (unsigned short*)d_ws;                       // 512 KB
  float* hb = (float*)((char*)d_ws +
                       (size_t)DEC_DIM * ENC_DIM * sizeof(unsigned short)); // 64 KB

  const size_t lds_bytes = (size_t)ROWS * LDS_STRIDE * sizeof(unsigned short)
                         + 8 * ROWS * sizeof(float);                // 66560 + 2048

  cvt_we_kernel<<<(DEC_DIM * ENC_DIM) / 256, 256, 0, stream>>>(W, Wt);
  hb_kernel<<<(B_DIM * DEC_DIM) / 256, 256, 0, stream>>>(dh, W, bias, hb);
  attn_scores_kernel<<<dim3(S_DIM / ROWS, B_DIM), 256, lds_bytes, stream>>>(
      enc, Wt, hb, v, out);
  softmax_kernel<<<B_DIM, 256, 0, stream>>>(out);
}
